// ModernBertAttention_16776142258399
// MI455X (gfx1250) — compile-verified, matrix operand order checked by reading, not run on hardware
//
#include <hip/hip_runtime.h>

#define BATCH 2
#define SEQ   2048
#define HID   768
#define NH    12
#define HD    64
#define QKV_N 2304

typedef __attribute__((ext_vector_type(16))) __bf16 v16bf;
typedef __attribute__((ext_vector_type(8)))  float  v8f;
typedef unsigned int u32;
typedef __attribute__((ext_vector_type(4)))  u32    u32x4;
typedef __attribute__((ext_vector_type(4)))  int    v4i;

// Async global->LDS copy (GLOBAL_LOAD_ASYNC_TO_LDS_B128, ASYNCcnt) if this
// toolchain exposes the builtins; otherwise fall back to a sync VGPR copy.
#if defined(__HIP_DEVICE_COMPILE__) && \
    __has_builtin(__builtin_amdgcn_global_load_async_to_lds_b128) && \
    __has_builtin(__builtin_amdgcn_s_wait_asynccnt)
#define USE_ASYNC_LDS 1
#else
#define USE_ASYNC_LDS 0
#endif

union AF { v16bf v; u32x4 q[2]; };
union CF { v8f   v; float f[8]; };

__device__ __forceinline__ __bf16 f2bf(float f) {
  u32 u = __builtin_bit_cast(u32, f);
  u32 r = u + 0x7FFFu + ((u >> 16) & 1u);   // round-to-nearest-even
  unsigned short h = (unsigned short)(r >> 16);
  return __builtin_bit_cast(__bf16, h);
}

__device__ __forceinline__ v8f wmma_bf16(const AF& a, const AF& b, v8f c) {
  return __builtin_amdgcn_wmma_f32_16x16x32_bf16(false, a.v, false, b.v,
                                                 (short)0, c, false, false);
}

// A/B fragment for 16-bit 16x32 (K) layout: lanes 0-15 own row (lane&15),
// lane>>4 selects K-half; two 16B loads per fragment.
__device__ __forceinline__ AF frag_ld(const __bf16* rowptr, int kh) {
  AF f;
  f.q[0] = *(const u32x4*)(rowptr + kh);
  f.q[1] = *(const u32x4*)(rowptr + 16 + kh);
  return f;
}

#if USE_ASYNC_LDS
typedef __attribute__((address_space(1))) v4i* gas_v4i_ptr;  // global (AS1)
typedef __attribute__((address_space(3))) v4i* lds_v4i_ptr;  // LDS (AS3)

__device__ __forceinline__ void async_cp_b128(const void* gsrc, void* ldst) {
  // Generic->AS casts via integers: AS1 shares the generic representation;
  // a generic LDS pointer carries the LDS byte offset in its low 32 bits.
  __builtin_amdgcn_global_load_async_to_lds_b128(
      (gas_v4i_ptr)(unsigned long long)gsrc,
      (lds_v4i_ptr)(unsigned int)(unsigned long long)ldst,
      0, 0);
}
#endif

// ---------------------------------------------------------------------------
// Kernel 1: qkv = x @ Wqkv (bf16 WMMA, f32 acc) + RoPE epilogue.
// Writes Q (scaled by 0.125) and K as [b,h,s,d] bf16, V transposed [b,h,d,s].
// ---------------------------------------------------------------------------
__global__ __launch_bounds__(256)
void qkv_rope_kernel(const float* __restrict__ x, const float* __restrict__ Wqkv,
                     __bf16* __restrict__ Q, __bf16* __restrict__ K,
                     __bf16* __restrict__ V) {
  __shared__ __bf16 lds_a[128 * 32];
  __shared__ __bf16 lds_b[128 * 32];
  const int tid = threadIdx.x, lane = tid & 31, w = tid >> 5;
  const int wm = w & 3, wn = w >> 2;               // 4 M-waves x 2 N-waves
  const int bm = blockIdx.y * 128, bn = blockIdx.x * 128;
  const int lc = lane & 15, kh = (lane >> 4) * 8;

  CF acc[2][4];
#pragma unroll
  for (int mt = 0; mt < 2; ++mt)
#pragma unroll
    for (int nt = 0; nt < 4; ++nt)
#pragma unroll
      for (int i = 0; i < 8; ++i) acc[mt][nt].f[i] = 0.0f;

  for (int k0 = 0; k0 < HID; k0 += 32) {
    { // stage A: x[bm..+127][k0..+31] -> lds_a[m][k] bf16
      int row = tid >> 1, half = tid & 1;
      const float* src = x + (size_t)(bm + row) * HID + k0 + half * 16;
      __bf16* dst = lds_a + row * 32 + half * 16;
      __builtin_prefetch(src + 32, 0, 0);
#pragma unroll
      for (int i = 0; i < 16; ++i) dst[i] = f2bf(src[i]);
    }
    { // stage B transposed: Wqkv[k0..+31][bn..+127] -> lds_b[n][k] bf16
      int krow = tid >> 3, seg = tid & 7;
      const float* src = Wqkv + (size_t)(k0 + krow) * QKV_N + bn + seg * 16;
      __builtin_prefetch(src + 32 * QKV_N, 0, 0);
#pragma unroll
      for (int i = 0; i < 16; ++i) lds_b[(seg * 16 + i) * 32 + krow] = f2bf(src[i]);
    }
    __syncthreads();
    AF a[2], b[4];
#pragma unroll
    for (int mt = 0; mt < 2; ++mt) a[mt] = frag_ld(lds_a + (wm * 32 + mt * 16 + lc) * 32, kh);
#pragma unroll
    for (int nt = 0; nt < 4; ++nt) b[nt] = frag_ld(lds_b + (wn * 64 + nt * 16 + lc) * 32, kh);
#pragma unroll
    for (int mt = 0; mt < 2; ++mt)
#pragma unroll
      for (int nt = 0; nt < 4; ++nt)
        acc[mt][nt].v = wmma_bf16(a[mt], b[nt], acc[mt][nt].v);
    __syncthreads();
  }

  // Epilogue: wave covers 64 head-aligned columns of qkv.
  const int nb = bn + wn * 64;
  const int sec = nb / HID;                 // 0=q 1=k 2=v
  const int head = (nb % HID) / HD;
  const int hi8 = (lane >> 4) * 8;

  if (sec == 2) {
#pragma unroll
    for (int nt = 0; nt < 4; ++nt) {
      int d = nt * 16 + lc;
#pragma unroll
      for (int mt = 0; mt < 2; ++mt)
#pragma unroll
        for (int j = 0; j < 8; ++j) {
          int m = bm + wm * 32 + mt * 16 + j + hi8;
          int bb = m >> 11, s = m & (SEQ - 1);
          V[(size_t)((bb * NH + head) * HD + d) * SEQ + s] = f2bf(acc[mt][nt].f[j]);
        }
    }
  } else {
    __bf16* dst = (sec == 0) ? Q : K;
    const float qscale = (sec == 0) ? 0.125f : 1.0f;   // 1/sqrt(64) folded into q
#pragma unroll
    for (int nt = 0; nt < 2; ++nt) {                   // nt pairs with nt+2 (d, d+32)
      int dm = nt * 16 + lc;
      float freq = exp2f(-0.4152410118609203f * (float)dm); // 10000^(-dm/32)
#pragma unroll
      for (int mt = 0; mt < 2; ++mt)
#pragma unroll
        for (int j = 0; j < 8; ++j) {
          int m = bm + wm * 32 + mt * 16 + j + hi8;
          int bb = m >> 11, s = m & (SEQ - 1);
          float sn, cs;
          __sincosf((float)s * freq, &sn, &cs);
          float lo = acc[mt][nt].f[j], hi = acc[mt][nt + 2].f[j];
          float olo = (lo * cs - hi * sn) * qscale;
          float ohi = (hi * cs + lo * sn) * qscale;
          __bf16* rp = dst + (size_t)((bb * NH + head) * SEQ + s) * HD + dm;
          rp[0]  = f2bf(olo);
          rp[32] = f2bf(ohi);
        }
    }
  }
}

// ---------------------------------------------------------------------------
// Kernel 2: sliding-window flash attention. 1 wave = 16 queries; 5 fixed
// 32-key chunks cover [q0-64, q0+95]. P goes C-layout -> LDS -> A-fragment.
// ---------------------------------------------------------------------------
__global__ __launch_bounds__(128)
void attn_kernel(const __bf16* __restrict__ Q, const __bf16* __restrict__ K,
                 const __bf16* __restrict__ Vt, const int* __restrict__ pad,
                 __bf16* __restrict__ O) {
  __shared__ __bf16 lds_p[4][16 * 32];
  const int tid = threadIdx.x, lane = tid & 31, w = tid >> 5;
  const int lc = lane & 15, kh = (lane >> 4) * 8, hi8 = (lane >> 4) * 8;
  const int q0 = (blockIdx.x * 4 + w) * 16;
  const int h = blockIdx.y, b = blockIdx.z;
  const __bf16* Qp = Q  + (size_t)((b * NH + h) * SEQ + q0) * HD;
  const __bf16* Kp = K  + (size_t)((b * NH + h) * SEQ) * HD;
  const __bf16* Vp = Vt + (size_t)((b * NH + h) * HD) * SEQ;
  const int* pb = pad + b * SEQ;

  AF aq[2];
  aq[0] = frag_ld(Qp + lc * HD, kh);
  aq[1] = frag_ld(Qp + lc * HD + 32, kh);

  CF accO[4], zc;
#pragma unroll
  for (int i = 0; i < 8; ++i) zc.f[i] = 0.0f;
#pragma unroll
  for (int dt = 0; dt < 4; ++dt)
#pragma unroll
    for (int i = 0; i < 8; ++i) accO[dt].f[i] = 0.0f;

  float mrow[8], lrow[8];
#pragma unroll
  for (int j = 0; j < 8; ++j) { mrow[j] = -1e30f; lrow[j] = 0.0f; }

  for (int c = 0; c < 5; ++c) {          // uniform trip count (barrier-safe)
    const int kc = q0 - 64 + c * 32;

    AF kf[2][2];
#pragma unroll
    for (int t = 0; t < 2; ++t) {
      int key = kc + t * 16 + lc;
      int kcl = key < 0 ? 0 : (key > SEQ - 1 ? SEQ - 1 : key);
      const __bf16* rp = Kp + (size_t)kcl * HD;
      kf[t][0] = frag_ld(rp, kh);
      kf[t][1] = frag_ld(rp + 32, kh);
    }

    CF sc[2];
#pragma unroll
    for (int t = 0; t < 2; ++t) {
      sc[t].v = wmma_bf16(aq[0], kf[t][0], zc.v);
      sc[t].v = wmma_bf16(aq[1], kf[t][1], sc[t].v);
    }

    float p0[8], p1[8], alpha[8];
#pragma unroll
    for (int j = 0; j < 8; ++j) {
      int qrow = q0 + j + hi8;
      int k0i = kc + lc, k1i = kc + 16 + lc;
      float s0 = sc[0].f[j], s1 = sc[1].f[j];
      if (k0i < 0 || k0i >= SEQ) s0 = -1e30f;
      else {
        int d = qrow - k0i; d = d < 0 ? -d : d;
        if (d > 64) s0 -= 10000.0f;
        s0 -= (1.0f - (float)pb[k0i]) * 10000.0f;
      }
      if (k1i < 0 || k1i >= SEQ) s1 = -1e30f;
      else {
        int d = qrow - k1i; d = d < 0 ? -d : d;
        if (d > 64) s1 -= 10000.0f;
        s1 -= (1.0f - (float)pb[k1i]) * 10000.0f;
      }
      float mx = fmaxf(s0, s1);
#pragma unroll
      for (int off = 1; off < 16; off <<= 1) mx = fmaxf(mx, __shfl_xor(mx, off));
      float nm = fmaxf(mrow[j], mx);
      float al = __expf(mrow[j] - nm);
      mrow[j] = nm;  alpha[j] = al;
      float e0 = __expf(s0 - nm), e1 = __expf(s1 - nm);
      p0[j] = e0; p1[j] = e1;
      float rs = e0 + e1;
#pragma unroll
      for (int off = 1; off < 16; off <<= 1) rs += __shfl_xor(rs, off);
      lrow[j] = lrow[j] * al + rs;
    }
#pragma unroll
    for (int dt = 0; dt < 4; ++dt)
#pragma unroll
      for (int j = 0; j < 8; ++j) accO[dt].f[j] *= alpha[j];

    // P (C-layout) -> LDS [m][k] bf16, then reload as A-fragment.
#pragma unroll
    for (int j = 0; j < 8; ++j) {
      int m = j + hi8;
      lds_p[w][m * 32 + lc]      = f2bf(p0[j]);
      lds_p[w][m * 32 + 16 + lc] = f2bf(p1[j]);
    }
    __syncthreads();
    AF pf = frag_ld(&lds_p[w][lc * 32], kh);

    int run0 = kc + kh;       run0 = run0 < 0 ? 0 : (run0 > SEQ - 8 ? SEQ - 8 : run0);
    int run1 = kc + 16 + kh;  run1 = run1 < 0 ? 0 : (run1 > SEQ - 8 ? SEQ - 8 : run1);
#pragma unroll
    for (int dt = 0; dt < 4; ++dt) {
      const __bf16* vrow = Vp + (size_t)(dt * 16 + lc) * SEQ;
      AF vf;
      vf.q[0] = *(const u32x4*)(vrow + run0);
      vf.q[1] = *(const u32x4*)(vrow + run1);
      accO[dt].v = wmma_bf16(pf, vf, accO[dt].v);
    }
    __syncthreads();
  }

#pragma unroll
  for (int dt = 0; dt < 4; ++dt)
#pragma unroll
    for (int j = 0; j < 8; ++j) {
      int s = q0 + j + hi8;
      float o = accO[dt].f[j] / lrow[j];
      O[(size_t)(b * SEQ + s) * HID + h * HD + dt * 16 + lc] = f2bf(o);
    }
}

// ---------------------------------------------------------------------------
// Kernel 3: out = O @ Wo (A already bf16), f32 output.
// A-tile staging uses GLOBAL_LOAD_ASYNC_TO_LDS_B128 when available.
// ---------------------------------------------------------------------------
__global__ __launch_bounds__(256)
void out_proj_kernel(const __bf16* __restrict__ O, const float* __restrict__ Wo,
                     float* __restrict__ out) {
  __shared__ __bf16 lds_a[128 * 32];
  __shared__ __bf16 lds_b[128 * 32];
  const int tid = threadIdx.x, lane = tid & 31, w = tid >> 5;
  const int wm = w & 3, wn = w >> 2;
  const int bm = blockIdx.y * 128, bn = blockIdx.x * 128;
  const int lc = lane & 15, kh = (lane >> 4) * 8, hi8 = (lane >> 4) * 8;

  CF acc[2][4];
#pragma unroll
  for (int mt = 0; mt < 2; ++mt)
#pragma unroll
    for (int nt = 0; nt < 4; ++nt)
#pragma unroll
      for (int i = 0; i < 8; ++i) acc[mt][nt].f[i] = 0.0f;

  for (int k0 = 0; k0 < HID; k0 += 32) {
    { // stage A (already bf16): 32B per thread, async to LDS when possible
      int row = tid >> 1, half = tid & 1;
      const u32x4* src = (const u32x4*)(O + (size_t)(bm + row) * HID + k0 + half * 16);
      u32x4* dst = (u32x4*)(lds_a + row * 32 + half * 16);
#if USE_ASYNC_LDS
      async_cp_b128(src, dst);
      async_cp_b128(src + 1, dst + 1);
#else
      dst[0] = src[0]; dst[1] = src[1];
#endif
    }
    { // stage B transposed with f32->bf16
      int krow = tid >> 3, seg = tid & 7;
      const float* src = Wo + (size_t)(k0 + krow) * HID + bn + seg * 16;
#pragma unroll
      for (int i = 0; i < 16; ++i) lds_b[(seg * 16 + i) * 32 + krow] = f2bf(src[i]);
    }
#if USE_ASYNC_LDS
    __builtin_amdgcn_s_wait_asynccnt(0);
#endif
    __syncthreads();
    AF a[2], b[4];
#pragma unroll
    for (int mt = 0; mt < 2; ++mt) a[mt] = frag_ld(lds_a + (wm * 32 + mt * 16 + lc) * 32, kh);
#pragma unroll
    for (int nt = 0; nt < 4; ++nt) b[nt] = frag_ld(lds_b + (wn * 64 + nt * 16 + lc) * 32, kh);
#pragma unroll
    for (int mt = 0; mt < 2; ++mt)
#pragma unroll
      for (int nt = 0; nt < 4; ++nt)
        acc[mt][nt].v = wmma_bf16(a[mt], b[nt], acc[mt][nt].v);
    __syncthreads();
  }

#pragma unroll
  for (int mt = 0; mt < 2; ++mt)
#pragma unroll
    for (int nt = 0; nt < 4; ++nt)
#pragma unroll
      for (int j = 0; j < 8; ++j) {
        int m = bm + wm * 32 + mt * 16 + j + hi8;
        int n = bn + wn * 64 + nt * 16 + lc;
        out[(size_t)m * HID + n] = acc[mt][nt].f[j];
      }
}

// ---------------------------------------------------------------------------
extern "C" void kernel_launch(void* const* d_in, const int* in_sizes, int n_in,
                              void* d_out, int out_size, void* d_ws, size_t ws_size,
                              hipStream_t stream) {
  const float* x    = (const float*)d_in[0];
  const int*   pad  = (const int*)d_in[1];
  const float* Wqkv = (const float*)d_in[2];
  const float* Wo   = (const float*)d_in[3];
  float*       out  = (float*)d_out;

  const size_t QKV_BYTES = (size_t)BATCH * NH * SEQ * HD * sizeof(__bf16); // 6 MiB each
  char* ws = (char*)d_ws;
  __bf16* Qb = (__bf16*)(ws);
  __bf16* Kb = (__bf16*)(ws + QKV_BYTES);
  __bf16* Vb = (__bf16*)(ws + 2 * QKV_BYTES);   // transposed [b,h,d,s]
  __bf16* Ob = (__bf16*)(ws + 3 * QKV_BYTES);   // [b,s,hid]

  qkv_rope_kernel<<<dim3(QKV_N / 128, (BATCH * SEQ) / 128), 256, 0, stream>>>(
      x, Wqkv, Qb, Kb, Vb);
  attn_kernel<<<dim3(SEQ / 64, NH, BATCH), 128, 0, stream>>>(
      Qb, Kb, Vb, pad, Ob);
  out_proj_kernel<<<dim3(HID / 128, (BATCH * SEQ) / 128), 256, 0, stream>>>(
      Ob, Wo, out);
}
